// ComplexDeformableConv2d_61254823575874
// MI455X (gfx1250) — matrix operand order repaired and checked by reading, not compiled, not run on hardware
//
#include <hip/hip_runtime.h>
#include <math.h>

// ---------------- types ----------------
typedef __bf16 bf16_t;
typedef __attribute__((ext_vector_type(2)))  bf16_t bf16x2;
typedef __attribute__((ext_vector_type(4)))  bf16_t bf16x4;
typedef __attribute__((ext_vector_type(8)))  bf16_t bf16x8;
typedef __attribute__((ext_vector_type(16))) bf16_t v16bf;
typedef __attribute__((ext_vector_type(8)))  float  v8f;

#define BATCH 8
#define CH    64
#define HH    128
#define WW    128
#define K9    9
#define COUT  64
#define HW    (HH*WW)

// stride (in bf16 elements) of one pixel row in the LDS B-staging buffers.
// 72*2 = 144 bytes: multiple of 16 (b128 aligned), 36 dwords -> bank-spread.
#define BSTRIDE 72

// ---------------- helpers ----------------
__device__ __forceinline__ int iclamp(int v, int lo, int hi) {
    return v < lo ? lo : (v > hi ? hi : v);
}

// A-fragment (16x32 bf16, M x K): per lane, 16 contiguous K values
// starting at (kc*32 + half*16).
__device__ __forceinline__ v16bf load_a_frag(const bf16_t* p) {
    bf16x8 lo = *(const bf16x8*)(p);
    bf16x8 hi = *(const bf16x8*)(p + 8);
    return __builtin_shufflevector(lo, hi, 0,1,2,3,4,5,6,7,8,9,10,11,12,13,14,15);
}

// B-fragment (32x16 bf16, K x N): per lane, 16 contiguous K values
// starting at (kc*32 + half*16)  ->  two ds_load_b128.
__device__ __forceinline__ v16bf load_b_frag(const bf16_t* p) {
    bf16x8 lo = *(const bf16x8*)(p);
    bf16x8 hi = *(const bf16x8*)(p + 8);
    return __builtin_shufflevector(lo, hi, 0,1,2,3,4,5,6,7,8,9,10,11,12,13,14,15);
}

// ---------------- kernel 1: pack P = Wr+Wi, Q = Wr-Wi as bf16 [k][m][c] ----------------
__global__ __launch_bounds__(256) void prep_pq_kernel(const float* __restrict__ wr,
                                                      const float* __restrict__ wi,
                                                      bf16_t* __restrict__ P,
                                                      bf16_t* __restrict__ Q) {
    int idx = blockIdx.x * 256 + threadIdx.x;           // 9*64*64 = 36864
    if (idx >= K9 * COUT * CH) return;
    int k = idx / (COUT * CH);
    int m = (idx >> 6) & 63;
    int c = idx & 63;
    float r = wr[(m * CH + c) * K9 + k];
    float i = wi[(m * CH + c) * K9 + k];
    P[idx] = (bf16_t)(r + i);
    Q[idx] = (bf16_t)(r - i);
}

// ---------------- kernel 2: offset + magnitude convs (direct, LDS-tiled) ----------------
// One block = one (b, ho) row, 128 threads = 128 output pixels.
__global__ __launch_bounds__(128) void offmag_kernel(const float* __restrict__ xr,
                                                     const float* __restrict__ xi,
                                                     const float* __restrict__ w_off,
                                                     const float* __restrict__ w_mag,
                                                     float* __restrict__ dyO,
                                                     float* __restrict__ dxO,
                                                     float* __restrict__ mgO) {
    __shared__ float sXr[3][130];
    __shared__ float sXi[3][130];
    __shared__ float sM [3][130];
    __shared__ float sWA[18 * 9];   // w_off, real-channel c
    __shared__ float sWB[18 * 9];   // w_off, imag-channel 64+c
    __shared__ float sWM[9 * 9];    // w_mag, channel c

    const int b  = blockIdx.x >> 7;
    const int ho = blockIdx.x & 127;
    const int t  = threadIdx.x;      // wo

    float accO[18];
    float accM[9];
#pragma unroll
    for (int o = 0; o < 18; ++o) accO[o] = 0.f;
#pragma unroll
    for (int o = 0; o < 9;  ++o) accM[o] = 0.f;

    const float* xrB = xr + (size_t)b * CH * HW;
    const float* xiB = xi + (size_t)b * CH * HW;

    for (int c = 0; c < CH; ++c) {
        __syncthreads();
        // stage 3 input rows (with halo + zero padding) and |x|
        for (int e = t; e < 3 * 130; e += 128) {
            int row = e / 130, col = e % 130;
            int h = ho - 1 + row, w = col - 1;
            float vr = 0.f, vi = 0.f;
            if (h >= 0 && h < HH && w >= 0 && w < WW) {
                vr = xrB[c * HW + h * WW + w];
                vi = xiB[c * HW + h * WW + w];
            }
            sXr[row][col] = vr;
            sXi[row][col] = vi;
            sM [row][col] = sqrtf(vr * vr + vi * vi);
        }
        // stage this channel's weights
        for (int e = t; e < 405; e += 128) {
            if (e < 162) {
                int o = e / 9, tap = e % 9;
                sWA[e] = w_off[((o * 2 * CH) + c) * 9 + tap];
            } else if (e < 324) {
                int e2 = e - 162; int o = e2 / 9, tap = e2 % 9;
                sWB[e2] = w_off[((o * 2 * CH) + CH + c) * 9 + tap];
            } else {
                int e2 = e - 324; int o = e2 / 9, tap = e2 % 9;
                sWM[e2] = w_mag[(o * CH + c) * 9 + tap];
            }
        }
        __syncthreads();
#pragma unroll
        for (int tap = 0; tap < 9; ++tap) {
            const int ky = tap / 3, kx = tap % 3;
            const float xrv = sXr[ky][t + kx];
            const float xiv = sXi[ky][t + kx];
            const float mv  = sM [ky][t + kx];
#pragma unroll
            for (int o = 0; o < 18; ++o)
                accO[o] = fmaf(xrv, sWA[o * 9 + tap], fmaf(xiv, sWB[o * 9 + tap], accO[o]));
#pragma unroll
            for (int o = 0; o < 9; ++o)
                accM[o] = fmaf(mv, sWM[o * 9 + tap], accM[o]);
        }
    }

#pragma unroll
    for (int k = 0; k < K9; ++k) {
        int idx = ((b * K9 + k) * HH + ho) * WW + t;
        dyO[idx] = accO[2 * k];
        dxO[idx] = accO[2 * k + 1];
        mgO[idx] = 1.f / (1.f + expf(-accM[k]));    // sigmoid
    }
}

// ---------------- kernel 3: deformable sample + complex GEMM (WMMA bf16) ----------------
// One block = one (b, ho) row of 128 pixels. 256 threads = 8 wave32.
// Wave (mTile, nHalf): mTile = wid&3 covers M rows [16*mTile,16*mTile+16),
// nHalf = wid>>2 covers pixels [64*nHalf, 64*nHalf+64) as 4 N-subtiles.
__global__ __launch_bounds__(256) void cdeform_main_kernel(const float* __restrict__ xr,
                                                           const float* __restrict__ xi,
                                                           const float* __restrict__ b_real,
                                                           const float* __restrict__ b_imag,
                                                           const float* __restrict__ dyI,
                                                           const float* __restrict__ dxI,
                                                           const float* __restrict__ mgI,
                                                           const bf16_t* __restrict__ P,
                                                           const bf16_t* __restrict__ Q,
                                                           float* __restrict__ out) {
    __shared__ __align__(16) bf16_t uLds[128 * BSTRIDE];  // u = sr - si, [pix][c]
    __shared__ __align__(16) bf16_t vLds[128 * BSTRIDE];  // v = sr + si, [pix][c]
    __shared__ int   sIdx[4][128];                        // clipped corner indices
    __shared__ float sW  [4][128];                        // corner weight * mag * valid

    const int b    = blockIdx.x >> 7;
    const int ho   = blockIdx.x & 127;
    const int t    = threadIdx.x;
    const int lane = t & 31;
    const int wid  = t >> 5;
    const int mTile = wid & 3;
    const int nHalf = wid >> 2;
    const int nOff  = lane & 15;
    const int half  = lane >> 4;

    v8f accP[4], accQ[4];
#pragma unroll
    for (int ns = 0; ns < 4; ++ns) { accP[ns] = (v8f)(0.f); accQ[ns] = (v8f)(0.f); }

    const float* xrB = xr + (size_t)b * CH * HW;
    const float* xiB = xi + (size_t)b * CH * HW;
    const int mrow = mTile * 16 + nOff;                   // A-fragment M row

    // gather-stage mapping: one pixel per thread, 32-channel strip
    const int gPix   = t & 127;
    const int gCBase = (t >> 7) * 32;

    for (int k = 0; k < K9; ++k) {
        __syncthreads();                                   // LDS reuse fence
        if (t < 128) {                                     // per-pixel bilinear params
            const int wo  = t;
            const int idx = ((b * K9 + k) * HH + ho) * WW + wo;
            const float py = (float)(ho - 1 + k / 3) + dyI[idx];
            const float px = (float)(wo - 1 + k % 3) + dxI[idx];
            const float m  = mgI[idx];
            const float y0f = floorf(py), x0f = floorf(px);
            const int y0 = (int)y0f, x0 = (int)x0f;
            const float wy1 = py - y0f, wx1 = px - x0f;
            const float wy[2] = {1.f - wy1, wy1};
            const float wx[2] = {1.f - wx1, wx1};
            const int   ys[2] = {y0, y0 + 1};
            const int   xs[2] = {x0, x0 + 1};
#pragma unroll
            for (int cy = 0; cy < 2; ++cy)
#pragma unroll
                for (int cx = 0; cx < 2; ++cx) {
                    const int yy = ys[cy], xx = xs[cx];
                    const bool ok = (yy >= 0) & (yy < HH) & (xx >= 0) & (xx < WW);
                    sIdx[cy * 2 + cx][wo] = iclamp(yy, 0, HH - 1) * WW + iclamp(xx, 0, WW - 1);
                    sW  [cy * 2 + cx][wo] = ok ? wy[cy] * wx[cx] * m : 0.f;
                }
        }
        __syncthreads();

        // gather + stage u/v into LDS. Params loaded once per thread; 4-channel
        // packed bf16x4 stores (ds_store_b64). x (67 MB) is fully L2-resident.
        {
            const int   id0 = sIdx[0][gPix], id1 = sIdx[1][gPix];
            const int   id2 = sIdx[2][gPix], id3 = sIdx[3][gPix];
            const float w0  = sW[0][gPix],   w1  = sW[1][gPix];
            const float w2  = sW[2][gPix],   w3  = sW[3][gPix];
            const float* prB = xrB + (size_t)gCBase * HW;
            const float* piB = xiB + (size_t)gCBase * HW;
            bf16_t* uRow = &uLds[gPix * BSTRIDE + gCBase];
            bf16_t* vRow = &vLds[gPix * BSTRIDE + gCBase];
#pragma unroll 2
            for (int cc = 0; cc < 32; cc += 4) {
                float uv[4], vv[4];
#pragma unroll
                for (int j = 0; j < 4; ++j) {
                    const float* pr = prB + (size_t)(cc + j) * HW;
                    const float* pi = piB + (size_t)(cc + j) * HW;
                    float sr = w0 * pr[id0];
                    sr = fmaf(w1, pr[id1], sr);
                    sr = fmaf(w2, pr[id2], sr);
                    sr = fmaf(w3, pr[id3], sr);
                    float si = w0 * pi[id0];
                    si = fmaf(w1, pi[id1], si);
                    si = fmaf(w2, pi[id2], si);
                    si = fmaf(w3, pi[id3], si);
                    uv[j] = sr - si;
                    vv[j] = sr + si;
                }
                bf16x4 up = {(bf16_t)uv[0], (bf16_t)uv[1], (bf16_t)uv[2], (bf16_t)uv[3]};
                bf16x4 vp = {(bf16_t)vv[0], (bf16_t)vv[1], (bf16_t)vv[2], (bf16_t)vv[3]};
                *(bf16x4*)(uRow + cc) = up;
                *(bf16x4*)(vRow + cc) = vp;
            }
        }
        __syncthreads();

        // GEMM over this k-chunk: K = 64 channels = 2 WMMA steps of K=32.
        // Load ALL B fragments for this K-chunk first, then issue the 8 WMMAs
        // back-to-back so DS latency is covered by one descending wait chain.
#pragma unroll
        for (int kc = 0; kc < 2; ++kc) {
            const bf16_t* pA = P + ((k * COUT + mrow) * CH) + kc * 32 + half * 16;
            const bf16_t* qA = Q + ((k * COUT + mrow) * CH) + kc * 32 + half * 16;
            const v16bf aP = load_a_frag(pA);
            const v16bf aQ = load_a_frag(qA);
            v16bf uB[4], vB[4];
#pragma unroll
            for (int ns = 0; ns < 4; ++ns) {
                const int pix = nHalf * 64 + ns * 16 + nOff;
                const int off = pix * BSTRIDE + kc * 32 + half * 16;
                uB[ns] = load_b_frag(&uLds[off]);
                vB[ns] = load_b_frag(&vLds[off]);
            }
#pragma unroll
            for (int ns = 0; ns < 4; ++ns) {
                accP[ns] = __builtin_amdgcn_wmma_f32_16x16x32_bf16(
                    false, aP, false, uB[ns], (short)0, accP[ns], false, false);
                accQ[ns] = __builtin_amdgcn_wmma_f32_16x16x32_bf16(
                    false, aQ, false, vB[ns], (short)0, accQ[ns], false, false);
            }
        }
    }

    // epilogue: real = (P.u + Q.v)/2 + (br-bi), imag = (P.u - Q.v)/2 + (bi-br)
    // C/D layout: VGPR r -> M = r + 8*half (lanes>=16), N = lane%16
    float brv[8], biv[8];
#pragma unroll
    for (int r = 0; r < 8; ++r) {
        const int m = mTile * 16 + r + half * 8;
        brv[r] = b_real[m];
        biv[r] = b_imag[m];
    }
#pragma unroll
    for (int ns = 0; ns < 4; ++ns) {
        const int wo = nHalf * 64 + ns * 16 + nOff;
#pragma unroll
        for (int r = 0; r < 8; ++r) {
            const int m  = mTile * 16 + r + half * 8;
            const float p  = accP[ns][r];
            const float q  = accQ[ns][r];
            const float re = 0.5f * (p + q) + (brv[r] - biv[r]);
            const float im = 0.5f * (p - q) + (biv[r] - brv[r]);
            float2 v2 = make_float2(re, im);
            *(float2*)(out + (size_t)((((b * COUT + m) * HH + ho) * WW + wo)) * 2) = v2;
        }
    }
}

// ---------------- launcher ----------------
extern "C" void kernel_launch(void* const* d_in, const int* in_sizes, int n_in,
                              void* d_out, int out_size, void* d_ws, size_t ws_size,
                              hipStream_t stream) {
    (void)in_sizes; (void)n_in; (void)out_size; (void)ws_size;
    const float* xr     = (const float*)d_in[0];
    const float* xi     = (const float*)d_in[1];
    const float* w_off  = (const float*)d_in[2];
    const float* w_mag  = (const float*)d_in[3];
    const float* w_real = (const float*)d_in[4];
    const float* w_imag = (const float*)d_in[5];
    const float* b_real = (const float*)d_in[6];
    const float* b_imag = (const float*)d_in[7];
    float* out = (float*)d_out;

    // workspace layout
    const size_t nOff = (size_t)BATCH * K9 * HW;   // 1,179,648 floats each
    float* dyW = (float*)d_ws;
    float* dxW = dyW + nOff;
    float* mgW = dxW + nOff;
    bf16_t* Pw = (bf16_t*)(mgW + nOff);            // 9*64*64 bf16
    bf16_t* Qw = Pw + (size_t)K9 * COUT * CH;

    prep_pq_kernel<<<(K9 * COUT * CH + 255) / 256, 256, 0, stream>>>(w_real, w_imag, Pw, Qw);
    offmag_kernel<<<BATCH * HH, 128, 0, stream>>>(xr, xi, w_off, w_mag, dyW, dxW, mgW);
    cdeform_main_kernel<<<BATCH * HH, 256, 0, stream>>>(xr, xi, b_real, b_imag,
                                                        dyW, dxW, mgW, Pw, Qw, out);
}
